// GATv2ClassificationNoEdgeAttr_78314433675487
// MI455X (gfx1250) — matrix operand order repaired and checked by reading, not compiled, not run on hardware
//
#include <hip/hip_runtime.h>
#include <hip/hip_bf16.h>

typedef __attribute__((ext_vector_type(4)))  __bf16 v4bf;
typedef __attribute__((ext_vector_type(8)))  __bf16 v8bf;
typedef __attribute__((ext_vector_type(16))) __bf16 v16bf;
typedef __attribute__((ext_vector_type(8)))  float  v8f;
typedef int v4i_vs __attribute__((vector_size(16)));

#define AS1 __attribute__((address_space(1)))
#define AS3 __attribute__((address_space(3)))

#if defined(__has_builtin)
#if __has_builtin(__builtin_amdgcn_global_load_async_to_lds_b128)
#define HAVE_ASYNC_LDS 1
#endif
#if __has_builtin(__builtin_amdgcn_s_wait_asynccnt)
#define HAVE_WAIT_ASYNC 1
#endif
#endif

__device__ __forceinline__ void wait_asynccnt0() {
#if defined(HAVE_WAIT_ASYNC)
    __builtin_amdgcn_s_wait_asynccnt(0);
#elif defined(HAVE_ASYNC_LDS)
    asm volatile("s_wait_asynccnt 0" ::: "memory");
#endif
}

// ---------------------------------------------------------------------------
// D[M, COLT*16] = A[M,128](bf16) @ Wt[COLT*16,128](bf16, pre-transposed) + bias
// Weight tile staged once per workgroup into LDS via async global->LDS copies
// (ASYNCcnt path); one wave32 per 16-row strip; COLT 16x16 tiles per wave.
// ---------------------------------------------------------------------------
template<int COLT>
__global__ __launch_bounds__(256) void k_gemm_wmma_bias(
    const __bf16* __restrict__ A, const __bf16* __restrict__ Wt,
    const float* __restrict__ bias, float* __restrict__ D, int M)
{
    __shared__ __bf16 sW[COLT * 16 * 128];   // COLT*4 KB

    // ---- cooperative staging: COLT*256 x 16B transfers, all 256 threads ----
    {
#if defined(HAVE_ASYNC_LDS)
        AS1 v4i_vs* g = (AS1 v4i_vs*)Wt;   // generic->global cast (drops const)
        AS3 v4i_vs* l = (AS3 v4i_vs*)sW;   // generic->LDS cast
        #pragma unroll
        for (int j = 0; j < COLT; ++j) {
            const int idx = (int)threadIdx.x + j * 256;
            __builtin_amdgcn_global_load_async_to_lds_b128(g + idx, l + idx, 0, 0);
        }
        wait_asynccnt0();
#else
        #pragma unroll
        for (int j = 0; j < COLT; ++j) {
            const int idx = (int)threadIdx.x + j * 256;
            ((v8bf*)sW)[idx] = ((const v8bf*)Wt)[idx];
        }
#endif
        __syncthreads();
    }

    const int lane = threadIdx.x & 31;
    const int wave = blockIdx.x * (blockDim.x >> 5) + (threadIdx.x >> 5);
    if (wave >= (M >> 4)) return;
    const int m0 = wave << 4;

    // A frag: lanes 0-15 -> row m0+lane, K 0..15; lanes 16-31 -> K 16..31
    const int aRow = m0 + (lane & 15);
    const int kA   = (lane < 16) ? 0 : 16;
    // B frag: lanes 0-15 -> col tile base + lane, K 0..15 ; lanes 16-31 -> K 16..31
    const int kB   = (lane < 16) ? 0 : 16;
    const int nc   = lane & 15;

    v8f acc[COLT];
    #pragma unroll
    for (int t = 0; t < COLT; ++t) acc[t] = (v8f){};

    const __bf16* arow = A + (size_t)aRow * 128 + kA;
    #pragma unroll
    for (int k0 = 0; k0 < 128; k0 += 32) {
        const v8bf alo = *(const v8bf*)(arow + k0);
        const v8bf ahi = *(const v8bf*)(arow + k0 + 8);
        const v16bf av = __builtin_shufflevector(
            alo, ahi, 0, 1, 2, 3, 4, 5, 6, 7, 8, 9, 10, 11, 12, 13, 14, 15);
        #pragma unroll
        for (int t = 0; t < COLT; ++t) {
            const v16bf bv =
                *(const v16bf*)(sW + (size_t)(t * 16 + nc) * 128 + k0 + kB);
            acc[t] = __builtin_amdgcn_wmma_f32_16x16x32_bf16(
                /*neg_a=*/false, av, /*neg_b=*/false, bv,
                /*c_mod=*/(short)0, acc[t], /*reuse_a=*/false, /*reuse_b=*/false);
        }
    }

    // D layout: lanes 0-15 rows 0..7 of column, lanes 16-31 rows 8..15
    const int r0 = (lane < 16) ? 0 : 8;
    #pragma unroll
    for (int t = 0; t < COLT; ++t) {
        const int col = t * 16 + nc;
        const float bz = bias[col];
        #pragma unroll
        for (int r = 0; r < 8; ++r)
            D[(size_t)(m0 + r0 + r) * (COLT * 16) + col] = acc[t][r] + bz;
    }
}

// ---------------------------------------------------------------------------
// Wt[n*128 + k] = (bf16) W[k*Ncols + n]   (transpose + convert, tiny)
// ---------------------------------------------------------------------------
__global__ void k_wt_bf16(const float* __restrict__ W, __bf16* __restrict__ Wt,
                          int Ncols)
{
    const int i = blockIdx.x * blockDim.x + threadIdx.x;  // over 128*Ncols
    if (i >= 128 * Ncols) return;
    const int k = i & 127, n = i >> 7;
    Wt[(size_t)n * 128 + k] = (__bf16)W[(size_t)k * Ncols + n];
}

// ---------------------------------------------------------------------------
// bf16 copy of an f32 tensor (8 elements per thread)
// ---------------------------------------------------------------------------
__global__ void k_f32_to_bf16(const float* __restrict__ in,
                              __bf16* __restrict__ out, int total8)
{
    const int i = blockIdx.x * blockDim.x + threadIdx.x;
    if (i >= total8) return;
    const float4 a = ((const float4*)in)[i * 2 + 0];
    const float4 b = ((const float4*)in)[i * 2 + 1];
    v8bf r;
    r[0] = (__bf16)a.x; r[1] = (__bf16)a.y; r[2] = (__bf16)a.z; r[3] = (__bf16)a.w;
    r[4] = (__bf16)b.x; r[5] = (__bf16)b.y; r[6] = (__bf16)b.z; r[7] = (__bf16)b.w;
    ((v8bf*)out)[i] = r;
}

// ---------------------------------------------------------------------------
// LayerNorm over F=128, one wave per node; writes bf16 (GEMM A operand)
// ---------------------------------------------------------------------------
__global__ __launch_bounds__(256) void k_layernorm_bf16(
    const float* __restrict__ h, const float* __restrict__ g,
    const float* __restrict__ b, __bf16* __restrict__ o, int nnodes)
{
    const int lane = threadIdx.x & 31;
    const int node = blockIdx.x * (blockDim.x >> 5) + (threadIdx.x >> 5);
    if (node >= nnodes) return;
    const float4 v = ((const float4*)(h + (size_t)node * 128))[lane];
    float s = v.x + v.y + v.z + v.w;
    float q = v.x * v.x + v.y * v.y + v.z * v.z + v.w * v.w;
    #pragma unroll
    for (int off = 16; off > 0; off >>= 1) {
        s += __shfl_xor(s, off, 32);
        q += __shfl_xor(q, off, 32);
    }
    const float mu  = s * (1.0f / 128.0f);
    const float var = q * (1.0f / 128.0f) - mu * mu;
    const float rs  = rsqrtf(var + 1e-5f);
    const float4 gg = ((const float4*)g)[lane];
    const float4 bb = ((const float4*)b)[lane];
    v4bf r;
    r[0] = (__bf16)((v.x - mu) * rs * gg.x + bb.x);
    r[1] = (__bf16)((v.y - mu) * rs * gg.y + bb.y);
    r[2] = (__bf16)((v.z - mu) * rs * gg.z + bb.z);
    r[3] = (__bf16)((v.w - mu) * rs * gg.w + bb.w);
    ((v4bf*)(o + (size_t)node * 128))[lane] = r;
}

// ---------------------------------------------------------------------------
// Zero accumulators, -inf max, zero denominators
// ---------------------------------------------------------------------------
__global__ void k_init(float* __restrict__ out, float* __restrict__ mx,
                       float* __restrict__ den, int n128, int n4)
{
    const int i = blockIdx.x * blockDim.x + threadIdx.x;
    if (i < n128) out[i] = 0.0f;
    if (i < n4) { mx[i] = -__builtin_huge_valf(); den[i] = 0.0f; }
}

// ---------------------------------------------------------------------------
// Pass 1: logits[e][h] = att[h] . leaky_relu(x_l[src] + x_r[dst]); atomicMax
// ---------------------------------------------------------------------------
__global__ __launch_bounds__(256) void k_edge_logits(
    const int* __restrict__ src, const int* __restrict__ dst, int E, int ET,
    const float* __restrict__ xl, const float* __restrict__ xr,
    const float* __restrict__ att, float* __restrict__ logits,
    float* __restrict__ mx)
{
    const int lane = threadIdx.x & 31;
    const int e = blockIdx.x * (blockDim.x >> 5) + (threadIdx.x >> 5);
    if (e >= ET) return;
    const int s = (e < E) ? src[e] : (e - E);
    const int d = (e < E) ? dst[e] : (e - E);
    const float4 a = ((const float4*)(xl + (size_t)s * 128))[lane];
    const float4 r = ((const float4*)(xr + (size_t)d * 128))[lane];
    float4 m;
    m.x = a.x + r.x; m.y = a.y + r.y; m.z = a.z + r.z; m.w = a.w + r.w;
    m.x = (m.x > 0.0f) ? m.x : 0.2f * m.x;
    m.y = (m.y > 0.0f) ? m.y : 0.2f * m.y;
    m.z = (m.z > 0.0f) ? m.z : 0.2f * m.z;
    m.w = (m.w > 0.0f) ? m.w : 0.2f * m.w;
    const int head = lane >> 3;
    const float4 at = ((const float4*)(att + head * 32))[lane & 7];
    float p = m.x * at.x + m.y * at.y + m.z * at.z + m.w * at.w;
    p += __shfl_xor(p, 1, 32);
    p += __shfl_xor(p, 2, 32);
    p += __shfl_xor(p, 4, 32);
    if ((lane & 7) == 0) {
        logits[(size_t)e * 4 + head] = p;
        atomicMax(&mx[(size_t)d * 4 + head], p);
    }
}

// ---------------------------------------------------------------------------
// Pass 2: a = exp(logit - mx[dst]); den[dst] += a
// ---------------------------------------------------------------------------
__global__ void k_edge_exp(const int* __restrict__ dst, int E, int ET,
                           float* __restrict__ logits,
                           const float* __restrict__ mx,
                           float* __restrict__ den)
{
    const int idx = blockIdx.x * blockDim.x + threadIdx.x;
    if (idx >= ET * 4) return;
    const int e = idx >> 2, hh = idx & 3;
    const int d = (e < E) ? dst[e] : (e - E);
    const float a = expf(logits[idx] - mx[(size_t)d * 4 + hh]);
    logits[idx] = a;
    atomicAdd(&den[(size_t)d * 4 + hh], a);
}

// ---------------------------------------------------------------------------
// Pass 3: out[dst] += x_l[src] * (a / den[dst])
// ---------------------------------------------------------------------------
__global__ __launch_bounds__(256) void k_edge_accum(
    const int* __restrict__ src, const int* __restrict__ dst, int E, int ET,
    const float* __restrict__ xl, const float* __restrict__ logits,
    const float* __restrict__ den, float* __restrict__ out)
{
    const int lane = threadIdx.x & 31;
    const int e = blockIdx.x * (blockDim.x >> 5) + (threadIdx.x >> 5);
    if (e >= ET) return;
    const int s = (e < E) ? src[e] : (e - E);
    const int d = (e < E) ? dst[e] : (e - E);
    const int head = lane >> 3;
    const float alpha = logits[(size_t)e * 4 + head] / den[(size_t)d * 4 + head];
    const float4 a = ((const float4*)(xl + (size_t)s * 128))[lane];
    float* o = out + (size_t)d * 128 + lane * 4;
    atomicAdd(o + 0, a.x * alpha);
    atomicAdd(o + 1, a.y * alpha);
    atomicAdd(o + 2, a.z * alpha);
    atomicAdd(o + 3, a.w * alpha);
}

// ---------------------------------------------------------------------------
// h = relu(out + gb) + h
// ---------------------------------------------------------------------------
__global__ void k_relu_res(float* __restrict__ h, const float* __restrict__ out,
                           const float* __restrict__ gb, int total)
{
    const int i = blockIdx.x * blockDim.x + threadIdx.x;
    if (i >= total) return;
    const float v = out[i] + gb[i & 127];
    h[i] = fmaxf(v, 0.0f) + h[i];
}

extern "C" void kernel_launch(void* const* d_in, const int* in_sizes, int n_in,
                              void* d_out, int out_size, void* d_ws, size_t ws_size,
                              hipStream_t stream) {
    (void)n_in; (void)out_size; (void)ws_size;
    const float* x    = (const float*)d_in[0];
    const int*   ei   = (const int*)d_in[1];
    const float* Wp   = (const float*)d_in[2];
    const float* bp   = (const float*)d_in[3];
    const float* ln_g = (const float*)d_in[4];
    const float* ln_b = (const float*)d_in[5];
    const float* Wl   = (const float*)d_in[6];
    const float* bl   = (const float*)d_in[7];
    const float* Wr   = (const float*)d_in[8];
    const float* br   = (const float*)d_in[9];
    const float* att  = (const float*)d_in[10];
    const float* gb   = (const float*)d_in[11];
    const float* Wc   = (const float*)d_in[12];
    const float* bc   = (const float*)d_in[13];

    const int F = 128, H = 4, K = 16, L = 2;
    const int N  = in_sizes[0] / F;      // 100000
    const int E  = in_sizes[1] / 2;      // 1600000
    const int ET = E + N;
    const int* srcI = ei;
    const int* dstI = ei + E;

    // ---- workspace carve-up (f32 region, then bf16 region) ----
    float* ws = (float*)d_ws;
    size_t off = 0;
    float* h   = ws + off; off += (size_t)N * F;
    float* out = ws + off; off += (size_t)N * F;
    float* xl  = ws + off; off += (size_t)N * F;
    float* xr  = ws + off; off += (size_t)N * F;
    float* lg  = ws + off; off += (size_t)ET * H;
    float* mx  = ws + off; off += (size_t)N * H;
    float* den = ws + off; off += (size_t)N * H;
    __bf16* ab  = (__bf16*)(ws + off);          // [N,128] bf16 GEMM A operand
    __bf16* wtb = ab + (size_t)N * F;           // transposed bf16 weights
    __bf16* WpT  = wtb;                         // 128*128
    __bf16* WlT0 = WpT  + 128 * 128;
    __bf16* WlT1 = WlT0 + 128 * 128;
    __bf16* WrT0 = WlT1 + 128 * 128;
    __bf16* WrT1 = WrT0 + 128 * 128;
    __bf16* WcT  = WrT1 + 128 * 128;            // 16*128

    const dim3 blk(256);
    auto gw = [](int waves) { return (unsigned)((waves + 7) / 8); };

    // ---- weight transpose/convert (tiny) ----
    k_wt_bf16<<<64, blk, 0, stream>>>(Wp, WpT, F);
    k_wt_bf16<<<64, blk, 0, stream>>>(Wl, WlT0, F);
    k_wt_bf16<<<64, blk, 0, stream>>>(Wl + (size_t)F * F, WlT1, F);
    k_wt_bf16<<<64, blk, 0, stream>>>(Wr, WrT0, F);
    k_wt_bf16<<<64, blk, 0, stream>>>(Wr + (size_t)F * F, WrT1, F);
    k_wt_bf16<<<8, blk, 0, stream>>>(Wc, WcT, K);

    const int rows16 = N / 16;                  // 6250 waves per GEMM
    const int cvt8   = N * F / 8;

    // h = x @ Wp + bp
    k_f32_to_bf16<<<(cvt8 + 255) / 256, blk, 0, stream>>>(x, ab, cvt8);
    k_gemm_wmma_bias<8><<<gw(rows16), blk, 0, stream>>>(ab, WpT, bp, h, N);

    for (int i = 0; i < L; ++i) {
        k_layernorm_bf16<<<gw(N), blk, 0, stream>>>(h, ln_g + i * F, ln_b + i * F,
                                                    ab, N);
        k_gemm_wmma_bias<8><<<gw(rows16), blk, 0, stream>>>(
            ab, (i == 0) ? WlT0 : WlT1, bl + i * F, xl, N);
        k_gemm_wmma_bias<8><<<gw(rows16), blk, 0, stream>>>(
            ab, (i == 0) ? WrT0 : WrT1, br + i * F, xr, N);
        k_init<<<(N * F + 255) / 256, blk, 0, stream>>>(out, mx, den, N * F, N * H);
        k_edge_logits<<<gw(ET), blk, 0, stream>>>(srcI, dstI, E, ET, xl, xr,
                                                  att + i * H * 32, lg, mx);
        k_edge_exp<<<(ET * 4 + 255) / 256, blk, 0, stream>>>(dstI, E, ET, lg, mx, den);
        k_edge_accum<<<gw(ET), blk, 0, stream>>>(srcI, dstI, E, ET, xl, lg, den, out);
        k_relu_res<<<(N * F + 255) / 256, blk, 0, stream>>>(h, out, gb + i * F, N * F);
    }

    // classification_output = h @ Wc + bc  -> d_out[0 : N*K]
    k_f32_to_bf16<<<(cvt8 + 255) / 256, blk, 0, stream>>>(h, ab, cvt8);
    k_gemm_wmma_bias<1><<<gw(rows16), blk, 0, stream>>>(ab, WcT, bc, (float*)d_out, N);
    // second tuple element: h -> d_out[N*K : N*K + N*F]
    (void)hipMemcpyAsync((float*)d_out + (size_t)N * K, h,
                         (size_t)N * F * sizeof(float),
                         hipMemcpyDeviceToDevice, stream);
}
